// VectorQuantizer_5927054868994
// MI455X (gfx1250) — matrix operand order repaired and checked by reading, not compiled, not run on hardware
//
#include <hip/hip_runtime.h>

// CDNA5 / gfx1250 VQ-VAE codebook quantizer.
// argmin_k ||z-e_k||^2 == argmin_k (||e_k||^2 - 2 z.e_k)   (row norm dropped)
// Cross term via fp32-emulating bf16x3 WMMA: z_hi*e_hi + z_hi*e_lo + z_lo*e_hi.
// Codebook staged in LDS via gfx1250 async global->LDS copies (ASYNCcnt).

typedef __attribute__((ext_vector_type(16))) __bf16 v16bf;
typedef __attribute__((ext_vector_type(8)))  float  v8f;

__device__ __forceinline__ v8f wmma_bf16(v16bf a, v16bf b, v8f c) {
    // (neg_a, A, neg_b, B, c_mod, C, reuse_a, reuse_b)
    return __builtin_amdgcn_wmma_f32_16x16x32_bf16(false, a, false, b,
                                                   (short)0, c, false, false);
}

// gfx1250 async copy: global -> LDS, 16B per lane, tracked on ASYNCcnt.
__device__ __forceinline__ void async_cp_b128(unsigned lds_off, const void* gaddr) {
    asm volatile("global_load_async_to_lds_b128 %0, %1, off"
                 :: "v"(lds_off), "v"((unsigned long long)gaddr) : "memory");
}
__device__ __forceinline__ void async_cp_b32(unsigned lds_off, const void* gaddr) {
    asm volatile("global_load_async_to_lds_b32 %0, %1, off"
                 :: "v"(lds_off), "v"((unsigned long long)gaddr) : "memory");
}
__device__ __forceinline__ void wait_async0() {
    asm volatile("s_wait_asynccnt 0x0" ::: "memory");
}

// Load 2x8 contiguous fp32 (k1..k1+7, k2..k2+7) and split into bf16 hi/lo fragments.
__device__ __forceinline__ void split16(const float* __restrict__ base, int k1, int k2,
                                        v16bf& h, v16bf& l) {
    float4 a = *(const float4*)(base + k1);
    float4 b = *(const float4*)(base + k1 + 4);
    float4 c = *(const float4*)(base + k2);
    float4 d = *(const float4*)(base + k2 + 4);
    float f[16] = {a.x, a.y, a.z, a.w, b.x, b.y, b.z, b.w,
                   c.x, c.y, c.z, c.w, d.x, d.y, d.z, d.w};
#pragma unroll
    for (int e = 0; e < 16; ++e) {
        __bf16 hh = (__bf16)f[e];
        h[e] = hh;
        l[e] = (__bf16)(f[e] - (float)hh);
    }
}

// ---------------- prep: split codebook to bf16 hi/lo + ||e||^2 -----------------
// grid = K blocks, block = 32 threads (1 wave). D fixed at 64.
__global__ void vq_prep_kernel(const float* __restrict__ w,
                               __bf16* __restrict__ bhi,
                               __bf16* __restrict__ blo,
                               float* __restrict__ wnorm) {
    const int k    = blockIdx.x;
    const int lane = threadIdx.x;          // 0..31
    float s = 0.0f;
#pragma unroll
    for (int t = 0; t < 2; ++t) {
        const int d = lane + t * 32;
        const float f = w[k * 64 + d];
        const __bf16 h = (__bf16)f;
        bhi[k * 64 + d] = h;
        blo[k * 64 + d] = (__bf16)(f - (float)h);
        s += f * f;
    }
#pragma unroll
    for (int m = 16; m >= 1; m >>= 1) s += __shfl_xor(s, m, 32);
    if (lane == 0) wnorm[k] = s;
}

// ---------------- main: per-wave 16-row tile vs all K codes --------------------
// block = 256 threads = 8 waves; each wave handles 16 rows of z.
// Dynamic LDS: bhi[K*64] bf16 | blo[K*64] bf16 | wnorm[K] f32  (130KB for K=512)
__global__ __launch_bounds__(256)
void vq_main_kernel(const float* __restrict__ x,
                    const float* __restrict__ w,
                    const __bf16* __restrict__ gbhi,
                    const __bf16* __restrict__ gblo,
                    const float* __restrict__ gwnorm,
                    float* __restrict__ outq,
                    int* __restrict__ outi,
                    int N, int K) {
    extern __shared__ char smem[];
    const int tid  = threadIdx.x;
    const int wave = tid >> 5;
    const int lane = tid & 31;

    // ---- stage codebook planes + norms into LDS via async copies (all threads)
    {
        const unsigned lbase   = (unsigned)(size_t)(&smem[0]);
        const unsigned plane   = (unsigned)K * 64u * 2u;     // bytes per bf16 plane
        const char* ghi = (const char*)gbhi;
        const char* glo = (const char*)gblo;
        const int chunks = (int)(plane / (256u * 16u));      // 16 for K=512
#pragma unroll
        for (int i = 0; i < 16; ++i) {
            if (i < chunks) {
                const unsigned off = ((unsigned)tid + (unsigned)i * 256u) * 16u;
                async_cp_b128(lbase + off,              ghi + off);
                async_cp_b128(lbase + plane + off,      glo + off);
            }
        }
        const unsigned nbase = lbase + 2u * plane;
#pragma unroll
        for (int i = 0; i < 2; ++i) {
            const int e = tid + i * 256;
            if (e < K) async_cp_b32(nbase + (unsigned)e * 4u,
                                    (const char*)gwnorm + (size_t)e * 4u);
        }
        wait_async0();
    }
    __syncthreads();

    const __bf16* sbhi = (const __bf16*)smem;
    const __bf16* sblo = sbhi + (size_t)K * 64;
    const float*  swn  = (const float*)(sblo + (size_t)K * 64);

    const int tile    = blockIdx.x * 8 + wave;
    const int rowBase = tile * 16;
    if (rowBase < N) {                     // wave-uniform
        // ---- A fragments: 16x32 bf16, contiguous k order (k = 16h + s), same as B.
        // lanes 0-15: M=lane,    K = {0..15} (+32 for chunk 1)
        // lanes 16-31: M=lane-16, K = {16..31}
        const int m = lane & 15;
        const float* xrow = x + (size_t)(rowBase + m) * 64;
        const int klo = (lane & 16) ? 16 : 0;

        v16bf ahi0, alo0, ahi1, alo1;
        split16(xrow, klo,      klo + 8,  ahi0, alo0);   // K chunk 0..31
        split16(xrow, klo + 32, klo + 40, ahi1, alo1);   // K chunk 32..63

        float best[8];
        int   bidx[8];
#pragma unroll
        for (int j = 0; j < 8; ++j) { best[j] = 3.4e38f; bidx[j] = 0; }

        // B fragment: 32x16 bf16 layout -> each lane = 16 contiguous K of one code.
        const int nb   = lane & 15;
        const int koff = (lane & 16) ? 16 : 0;
        const int numTiles = K >> 4;

        for (int ct = 0; ct < numTiles; ++ct) {
            const int n = ct * 16 + nb;
            const __bf16* bh = sbhi + n * 64 + koff;
            const __bf16* bl = sblo + n * 64 + koff;
            const v16bf bh0 = *(const v16bf*)(bh);        // K 0..31
            const v16bf bh1 = *(const v16bf*)(bh + 32);   // K 32..63
            const v16bf bl0 = *(const v16bf*)(bl);
            const v16bf bl1 = *(const v16bf*)(bl + 32);

            // two independent accumulator chains (chunk0 / chunk1) for WMMA ILP
            v8f accA = {}, accB = {};
            accA = wmma_bf16(ahi0, bh0, accA);   // hi*hi  chunk0
            accB = wmma_bf16(ahi1, bh1, accB);   // hi*hi  chunk1
            accA = wmma_bf16(ahi0, bl0, accA);   // hi*lo
            accB = wmma_bf16(ahi1, bl1, accB);
            accA = wmma_bf16(alo0, bh0, accA);   // lo*hi
            accB = wmma_bf16(alo1, bh1, accB);

            const float wn = swn[n];
#pragma unroll
            for (int j = 0; j < 8; ++j) {
                const float s = wn - 2.0f * (accA[j] + accB[j]);
                if (s < best[j]) { best[j] = s; bidx[j] = n; }  // asc n: ties keep first
            }
        }

        // ---- butterfly min-reduce within each 16-lane half (rows j / j+8)
#pragma unroll
        for (int j = 0; j < 8; ++j) {
#pragma unroll
            for (int msk = 1; msk <= 8; msk <<= 1) {
                const float od = __shfl_xor(best[j], msk, 32);
                const int   oi = __shfl_xor(bidx[j], msk, 32);
                if (od < best[j] || (od == best[j] && oi < bidx[j])) {
                    best[j] = od; bidx[j] = oi;
                }
            }
        }

        // ---- outputs: gather exact fp32 codebook rows + indices
#pragma unroll
        for (int j = 0; j < 8; ++j) {
            const int idxLo = __shfl(bidx[j], 0, 32);    // row rowBase + j
            const int idxHi = __shfl(bidx[j], 16, 32);   // row rowBase + j + 8
            const int r0 = rowBase + j;
            const int r1 = rowBase + j + 8;
            const float2 v0 = ((const float2*)(w + (size_t)idxLo * 64))[lane];
            const float2 v1 = ((const float2*)(w + (size_t)idxHi * 64))[lane];
            ((float2*)(outq + (size_t)r0 * 64))[lane] = v0;
            ((float2*)(outq + (size_t)r1 * 64))[lane] = v1;
            if (lane == 0) outi[r0] = idxLo;
            if (lane == 1) outi[r1] = idxHi;
        }
    }
}

extern "C" void kernel_launch(void* const* d_in, const int* in_sizes, int n_in,
                              void* d_out, int out_size, void* d_ws, size_t ws_size,
                              hipStream_t stream) {
    const float* x = (const float*)d_in[0];   // [N, 64] flat
    const float* w = (const float*)d_in[1];   // [K, 64]
    const int D = 64;
    const int N = in_sizes[0] / D;            // 131072
    const int K = in_sizes[1] / D;            // 512

    // workspace layout: bhi[K*64] bf16 | blo[K*64] bf16 | wnorm[K] f32
    __bf16* bhi   = (__bf16*)d_ws;
    __bf16* blo   = bhi + (size_t)K * D;
    float*  wnorm = (float*)(blo + (size_t)K * D);

    float* outq = (float*)d_out;
    int*   outi = (int*)d_out + (size_t)N * D;

    vq_prep_kernel<<<K, 32, 0, stream>>>(w, bhi, blo, wnorm);

    const int tiles  = N / 16;                              // 8192
    const int blocks = (tiles + 7) / 8;                     // 1024
    const size_t ldsBytes = (size_t)K * D * 2 * 2 + (size_t)K * 4;  // 130KB
    vq_main_kernel<<<blocks, 256, ldsBytes, stream>>>(x, w, bhi, blo, wnorm,
                                                      outq, outi, N, K);
}
